// MultiHeadAttention_22368189678341
// MI455X (gfx1250) — matrix operand order repaired and checked by reading, not compiled, not run on hardware
//
#include <hip/hip_runtime.h>

#define DEVINL __device__ __forceinline__

#if defined(__HIP_DEVICE_COMPILE__) && \
    __has_builtin(__builtin_amdgcn_global_load_async_to_lds_b128)
#define HAVE_ASYNC_LDS 1
#else
#define HAVE_ASYNC_LDS 0
#endif

namespace {

constexpr int kB     = 4;
constexpr int kS     = 2048;
constexpr int kEmbed = 1024;
constexpr int kHeads = 16;
constexpr int kHd    = 64;        // head dim
constexpr int kM     = kB * kS;   // 8192 tokens

typedef __attribute__((ext_vector_type(16))) __bf16 v16bf;
typedef __attribute__((ext_vector_type(8)))  __bf16 v8bf;
typedef __attribute__((ext_vector_type(8)))  float  v8f;
typedef int v4i __attribute__((vector_size(16)));   // matches builtin pointee

DEVINL v8bf ld8(const __bf16* p) { return *(const v8bf*)p; }

DEVINL v16bf cat16(v8bf lo, v8bf hi) {
  return __builtin_shufflevector(lo, hi, 0, 1, 2, 3, 4, 5, 6, 7,
                                         8, 9, 10, 11, 12, 13, 14, 15);
}

// 8 contiguous fp32 -> 8 bf16 (works for global or LDS source)
DEVINL v8bf cvt8(const float* p) {
  float4 f0 = *(const float4*)p;
  float4 f1 = *(const float4*)(p + 4);
  v8bf v;
  v[0] = (__bf16)f0.x; v[1] = (__bf16)f0.y; v[2] = (__bf16)f0.z; v[3] = (__bf16)f0.w;
  v[4] = (__bf16)f1.x; v[5] = (__bf16)f1.y; v[6] = (__bf16)f1.z; v[7] = (__bf16)f1.w;
  return v;
}

DEVINL v16bf ones16() {
  v16bf v;
#pragma unroll
  for (int i = 0; i < 16; ++i) v[i] = (__bf16)1.0f;
  return v;
}

DEVINL v8f wmma_bf16(v16bf a, v16bf b, v8f c) {
  // D = A(16x32) * B(32x16) + C, f32 accumulate
  return __builtin_amdgcn_wmma_f32_16x16x32_bf16(
      /*neg_a=*/false, a, /*neg_b=*/false, b,
      /*c_mod=*/(short)0, c, /*reuse_a=*/false, /*reuse_b=*/false);
}

#if HAVE_ASYNC_LDS
// one 16-byte async global->LDS transfer per lane (ASYNCcnt-tracked)
DEVINL void async_cp16(const void* g, void* l) {
  __builtin_amdgcn_global_load_async_to_lds_b128(
      (__attribute__((address_space(1))) v4i*)g,
      (__attribute__((address_space(3))) v4i*)l, 0, 0);
}
template <int N>
DEVINL void wait_async() {
#if __has_builtin(__builtin_amdgcn_s_wait_asynccnt)
  __builtin_amdgcn_s_wait_asynccnt(N);
#else
  asm volatile("s_wait_asynccnt %0" ::"i"(N));
#endif
}
#endif

// ---------------------------------------------------------------------------
// GEMM:  Out[M,N] = A[M,K] @ W[N,K]^T + bias   (torch Linear convention)
// A fp32 or bf16 (template), W/bias fp32, Out fp32 or bf16 (optionally V^T).
// 256 threads = 8 waves; tile 64x64; K-step 32.
// Async path: double-buffered GLOBAL_LOAD_ASYNC_TO_LDS_B128, raw dtype in
// LDS, convert to bf16 at fragment load. Fallback: sync staging.
// ---------------------------------------------------------------------------
constexpr int BM = 64, BN = 64, BK = 32, LDT = BK + 8;

template <bool A_BF16, bool OUT_F32, bool V_TRANS>
__global__ __launch_bounds__(256)
void gemm_bias_wmma(const void* __restrict__ Ap, const float* __restrict__ W,
                    const float* __restrict__ bias, void* __restrict__ Out) {
  const int tid  = threadIdx.x;
  const int wid  = tid >> 5;
  const int lane = tid & 31;
  const int hi   = lane >> 4;    // lane half (C rows r vs r+8; frag K halves)
  const int ln   = lane & 15;
  const int m0   = blockIdx.x * BM;
  const int n0   = blockIdx.y * BN;
  const int wm   = (wid & 3) * 16;   // wave's 16-row strip in tile
  const int wn   = (wid >> 2) * 32;  // wave's 32-col strip in tile
  const int lr   = tid >> 2;         // staging row 0..63
  const int lc   = (tid & 3) * 8;    // staging col chunk 0/8/16/24

  v8f acc0 = {};
  v8f acc1 = {};

#if HAVE_ASYNC_LDS
  // ---------------- async double-buffered pipeline ----------------
  constexpr int ROWB = 144;  // LDS row stride in bytes (16B aligned)
  __shared__ __align__(16) char sA[2 * BM * ROWB];
  __shared__ __align__(16) char sW[2 * BM * ROWB];
  constexpr int AES  = A_BF16 ? 2 : 4;         // staged A element size
  constexpr int TOPS = (A_BF16 ? 1 : 2) + 2;   // async ops per thread / tile

  auto issue = [&](int buf, int k0) {
    char* arow = &sA[(size_t)(buf * BM + lr) * ROWB];
    const char* ga =
        (const char*)Ap + ((size_t)(m0 + lr) * kEmbed + k0 + lc) * AES;
    async_cp16(ga, arow + lc * AES);
    if (!A_BF16) async_cp16(ga + 16, arow + lc * AES + 16);
    char* wrow = &sW[(size_t)(buf * BM + lr) * ROWB];
    const char* gw = (const char*)(W + (size_t)(n0 + lr) * kEmbed + k0 + lc);
    async_cp16(gw, wrow + lc * 4);
    async_cp16(gw + 16, wrow + lc * 4 + 16);
  };

  issue(0, 0);
  for (int k0 = 0; k0 < kEmbed; k0 += BK) {
    const int buf = (k0 >> 5) & 1;
    if (k0 + BK < kEmbed) {
      issue(buf ^ 1, k0 + BK);   // prefetch next tile into other buffer
      wait_async<TOPS>();        // only the just-issued ops may remain
    } else {
      wait_async<0>();
    }
    __syncthreads();

    // fragments (convert fp32->bf16 at load where needed)
    const char* arow = &sA[(size_t)(buf * BM + wm + ln) * ROWB];
    v16bf af;
    if (A_BF16) {
      const __bf16* ar = (const __bf16*)arow;
      af = cat16(ld8(ar + 16 * hi), ld8(ar + 16 * hi + 8));
    } else {
      const float* ar = (const float*)arow;
      af = cat16(cvt8(ar + 16 * hi), cvt8(ar + 16 * hi + 8));
    }
    const float* br0 = (const float*)&sW[(size_t)(buf * BM + wn + ln) * ROWB];
    v16bf bf0 = cat16(cvt8(br0 + 16 * hi), cvt8(br0 + 16 * hi + 8));
    const float* br1 =
        (const float*)&sW[(size_t)(buf * BM + wn + 16 + ln) * ROWB];
    v16bf bf1 = cat16(cvt8(br1 + 16 * hi), cvt8(br1 + 16 * hi + 8));

    acc0 = wmma_bf16(af, bf0, acc0);
    acc1 = wmma_bf16(af, bf1, acc1);
    __syncthreads();  // reads done before this buffer is overwritten
  }
#else
  // ---------------- synchronous staging fallback ----------------
  __shared__ __bf16 sA[BM][LDT];
  __shared__ __bf16 sB[BN][LDT];

  for (int k0 = 0; k0 < kEmbed; k0 += BK) {
    if (A_BF16) {
      const __bf16* A = (const __bf16*)Ap;
      *(v8bf*)&sA[lr][lc] = ld8(A + (size_t)(m0 + lr) * kEmbed + k0 + lc);
    } else {
      const float* A = (const float*)Ap;
      *(v8bf*)&sA[lr][lc] = cvt8(A + (size_t)(m0 + lr) * kEmbed + k0 + lc);
    }
    *(v8bf*)&sB[lr][lc] = cvt8(W + (size_t)(n0 + lr) * kEmbed + k0 + lc);

    if (k0 + BK < kEmbed) {
      const char* apf = (const char*)Ap +
          ((size_t)(m0 + lr) * kEmbed + k0 + BK + lc) * (A_BF16 ? 2 : 4);
      __builtin_prefetch(apf, 0, 1);
      __builtin_prefetch(W + (size_t)(n0 + lr) * kEmbed + k0 + BK + lc, 0, 1);
    }
    __syncthreads();

    const __bf16* ar = &sA[wm + ln][0];
    v16bf af = cat16(ld8(ar + 16 * hi), ld8(ar + 16 * hi + 8));
    const __bf16* br0 = &sB[wn + ln][0];
    v16bf bf0 = cat16(ld8(br0 + 16 * hi), ld8(br0 + 16 * hi + 8));
    const __bf16* br1 = &sB[wn + 16 + ln][0];
    v16bf bf1 = cat16(ld8(br1 + 16 * hi), ld8(br1 + 16 * hi + 8));

    acc0 = wmma_bf16(af, bf0, acc0);
    acc1 = wmma_bf16(af, bf1, acc1);
    __syncthreads();
  }
#endif

  // --- epilogue: bias + store.  C layout: row = r + 8*hi, col = ln ---
#pragma unroll
  for (int nt = 0; nt < 2; ++nt) {
    v8f acc = nt ? acc1 : acc0;
    const int col = n0 + wn + nt * 16 + ln;
    const float bv = bias[col];
#pragma unroll
    for (int r = 0; r < 8; ++r) {
      const int row = m0 + wm + r + 8 * hi;
      const float v = acc[r] + bv;
      if (OUT_F32) {
        ((float*)Out)[(size_t)row * kEmbed + col] = v;
      } else if (V_TRANS) {
        // store V transposed per head: [B, H, Dh, S] for contiguous PV B-frags
        const int bb = row >> 11, ss = row & (kS - 1);
        const int hh = col >> 6, dd = col & (kHd - 1);
        ((__bf16*)Out)[((size_t)((bb * kHeads + hh) * kHd + dd)) * kS + ss] =
            (__bf16)v;
      } else {
        ((__bf16*)Out)[(size_t)row * kEmbed + col] = (__bf16)v;
      }
    }
  }
}

// ---------------------------------------------------------------------------
// Flash attention, causal. One wave per 16-query tile; 32-key blocks.
// S = Q*K^T via 4 WMMA; row-max via ds_bpermute butterflies; P relayout
// through per-wave LDS (s_wait_dscnt intra-wave sync); row-sum via one extra
// WMMA against an all-ones B tile (C layout == per-lane lsum layout, zero
// shuffles); PV via 4 WMMA.
// ---------------------------------------------------------------------------
__global__ __launch_bounds__(256)
void flash_attn_wmma(const void* __restrict__ Qp, const void* __restrict__ Kp,
                     const void* __restrict__ Vp, void* __restrict__ Op) {
  const __bf16* Qb = (const __bf16*)Qp;
  const __bf16* Kb = (const __bf16*)Kp;
  const __bf16* Vt = (const __bf16*)Vp;   // [B, H, Dh, S]
  __bf16* Ob = (__bf16*)Op;               // [B*S, EMBED]

  __shared__ __bf16 sP[8][16][32];        // per-wave P tile (q x key), bf16

  const int tid  = threadIdx.x;
  const int wid  = tid >> 5;
  const int lane = tid & 31;
  const int hi   = lane >> 4;
  const int ln   = lane & 15;
  const int h    = blockIdx.y;
  const int b    = blockIdx.z;
  const int q0   = (blockIdx.x * 8 + wid) * 16;

  // Q tile 16x64 -> two A-frags (d 0..31, 32..63); rows = q0+ln
  const __bf16* qrow = Qb + ((size_t)(b * kS + q0 + ln)) * kEmbed + h * kHd;
  const v16bf aq0 = cat16(ld8(qrow + 16 * hi),      ld8(qrow + 16 * hi + 8));
  const v16bf aq1 = cat16(ld8(qrow + 32 + 16 * hi), ld8(qrow + 32 + 16 * hi + 8));
  const v16bf vones = ones16();

  v8f o[4] = {};
  float mr[8], lsum[8];
#pragma unroll
  for (int r = 0; r < 8; ++r) { mr[r] = -3.0e38f; lsum[r] = 0.f; }

  const float sc = 0.125f;                 // 1/sqrt(64)
  const int nkb = (q0 + 16 + 31) >> 5;     // key blocks up to causal diagonal

  for (int kb = 0; kb < nkb; ++kb) {
    const int kbase = kb * 32;

    // ---- S = Q K^T : two 16x16 C tiles (keys kbase..+15, kbase+16..+31) ----
    const __bf16* kr0 = Kb + ((size_t)(b * kS + kbase + ln)) * kEmbed + h * kHd;
    const __bf16* kr1 = kr0 + (size_t)16 * kEmbed;
    v8f s0 = {}, s1 = {};
    {
      v16bf b00 = cat16(ld8(kr0 + 16 * hi),      ld8(kr0 + 16 * hi + 8));
      v16bf b01 = cat16(ld8(kr0 + 32 + 16 * hi), ld8(kr0 + 32 + 16 * hi + 8));
      s0 = wmma_bf16(aq0, b00, s0);
      s0 = wmma_bf16(aq1, b01, s0);
      v16bf b10 = cat16(ld8(kr1 + 16 * hi),      ld8(kr1 + 16 * hi + 8));
      v16bf b11 = cat16(ld8(kr1 + 32 + 16 * hi), ld8(kr1 + 32 + 16 * hi + 8));
      s1 = wmma_bf16(aq0, b10, s1);
      s1 = wmma_bf16(aq1, b11, s1);
    }

    // ---- mask + scale + online max; write P = exp(s - m_new) to LDS ----
#pragma unroll
    for (int r = 0; r < 8; ++r) {
      const int qi = q0 + r + 8 * hi;
      const float f0 = (kbase + ln      <= qi) ? s0[r] * sc : -3.0e38f;
      const float f1 = (kbase + 16 + ln <= qi) ? s1[r] * sc : -3.0e38f;
      float rm = fmaxf(f0, f1);
      rm = fmaxf(rm, __shfl_xor(rm, 1, 32));
      rm = fmaxf(rm, __shfl_xor(rm, 2, 32));
      rm = fmaxf(rm, __shfl_xor(rm, 4, 32));
      rm = fmaxf(rm, __shfl_xor(rm, 8, 32));
      const float mnew = fmaxf(mr[r], rm);
      const float al = __expf(mr[r] - mnew);
      mr[r] = mnew;
      lsum[r] *= al;
      o[0][r] *= al; o[1][r] *= al; o[2][r] *= al; o[3][r] *= al;
      sP[wid][r + 8 * hi][ln]      = (__bf16)__expf(f0 - mnew);
      sP[wid][r + 8 * hi][16 + ln] = (__bf16)__expf(f1 - mnew);
    }
    // intra-wave LDS visibility: DS ops are in-order per wave; drain DScnt
    asm volatile("s_wait_dscnt 0" ::: "memory");
    __builtin_amdgcn_wave_barrier();

    // ---- P (16x32 A-frag) from LDS ----
    const __bf16* pr = &sP[wid][ln][0];
    const v16bf pf = cat16(ld8(pr + 16 * hi), ld8(pr + 16 * hi + 8));

    // row-sum of P via WMMA against ones: every C column = sum_k P[q,k]
    v8f rsum = {};
    rsum = wmma_bf16(pf, vones, rsum);

    // ---- PV accumulate over 4 N-tiles of Dh ----
#pragma unroll
    for (int nt = 0; nt < 4; ++nt) {
      const __bf16* vr = Vt +
          ((size_t)((b * kHeads + h) * kHd + nt * 16 + ln)) * kS + kbase +
          16 * hi;
      v16bf bv = cat16(ld8(vr), ld8(vr + 8));
      o[nt] = wmma_bf16(pf, bv, o[nt]);
    }
#pragma unroll
    for (int r = 0; r < 8; ++r) lsum[r] += rsum[r];
    __builtin_amdgcn_wave_barrier();
  }

  // ---- normalize and store attn output [B*S, EMBED] (bf16) ----
#pragma unroll
  for (int nt = 0; nt < 4; ++nt) {
    const int col = h * kHd + nt * 16 + ln;
#pragma unroll
    for (int r = 0; r < 8; ++r) {
      const int tok = b * kS + q0 + r + 8 * hi;
      Ob[(size_t)tok * kEmbed + col] = (__bf16)(o[nt][r] / lsum[r]);
    }
  }
}

}  // namespace

extern "C" void kernel_launch(void* const* d_in, const int* in_sizes, int n_in,
                              void* d_out, int out_size, void* d_ws,
                              size_t ws_size, hipStream_t stream) {
  (void)in_sizes; (void)n_in; (void)out_size; (void)ws_size;
  const float* query  = (const float*)d_in[0];
  const float* key_in = (const float*)d_in[1];
  const float* value  = (const float*)d_in[2];
  // d_in[3] = causal mask: known statically, unused
  const float* Wq = (const float*)d_in[4];
  const float* bq = (const float*)d_in[5];
  const float* Wk = (const float*)d_in[6];
  const float* bk = (const float*)d_in[7];
  const float* Wv = (const float*)d_in[8];
  const float* bv = (const float*)d_in[9];
  const float* Wo = (const float*)d_in[10];
  const float* bo = (const float*)d_in[11];

  // workspace: Q, K, V^T, attn — all bf16, 16 MB each (64 MB total)
  char* ws = (char*)d_ws;
  const size_t seg = (size_t)kM * kEmbed * 2;
  void* Qb = ws;
  void* Kb = ws + seg;
  void* Vt = ws + 2 * seg;
  void* At = ws + 3 * seg;

  dim3 gg(kM / BM, kEmbed / BN);
  dim3 bb(256);

  gemm_bias_wmma<false, false, false><<<gg, bb, 0, stream>>>(query,  Wq, bq, Qb);
  gemm_bias_wmma<false, false, false><<<gg, bb, 0, stream>>>(key_in, Wk, bk, Kb);
  gemm_bias_wmma<false, false, true ><<<gg, bb, 0, stream>>>(value,  Wv, bv, Vt);

  dim3 ga(kS / (16 * 8), kHeads, kB);  // (16, 16, 4): 8 query tiles per block
  flash_attn_wmma<<<ga, bb, 0, stream>>>(Qb, Kb, Vt, At);

  gemm_bias_wmma<true, true, false><<<gg, bb, 0, stream>>>(At, Wo, bo, d_out);
}
